// SSGSemanticSegmentation_55972013801668
// MI455X (gfx1250) — matrix operand order repaired and checked by reading, not compiled, not run on hardware
//
#include <hip/hip_runtime.h>
#include <hip/hip_bf16.h>
#include <math.h>

typedef __attribute__((ext_vector_type(16))) _Float16 v16h;
typedef __attribute__((ext_vector_type(8)))  _Float16 v8h;
typedef __attribute__((ext_vector_type(8)))  float    v8f;

#define NSAMPLE 50
#define BN_EPS  1e-5f

static inline int cdiv_i(int a, int b) { return (a + b - 1) / b; }
static inline int round32(int x) { return (x + 31) & ~31; }

// ---------------------------------------------------------------------------
// (B,6,N) -> xyz (B,N,3) and normals (B,N,3)
// ---------------------------------------------------------------------------
__global__ void split_kernel(const float* __restrict__ x, float* __restrict__ xyz,
                             float* __restrict__ nor, int B, int N) {
  int t = blockIdx.x * blockDim.x + threadIdx.x;
  if (t >= B * N) return;
  int b = t / N, n = t % N;
  const float* xb = x + (size_t)b * 6 * N;
  float* o1 = xyz + (size_t)t * 3;
  float* o2 = nor + (size_t)t * 3;
  o1[0] = xb[0 * N + n]; o1[1] = xb[1 * N + n]; o1[2] = xb[2 * N + n];
  o2[0] = xb[3 * N + n]; o2[1] = xb[4 * N + n]; o2[2] = xb[5 * N + n];
}

// ---------------------------------------------------------------------------
// Farthest point sampling: one 256-thread block per batch, dist in LDS.
// Emits OLD farthest, then updates dist, then argmax (matches the scan).
// ---------------------------------------------------------------------------
__global__ void fps_kernel(const float* __restrict__ xyz, int* __restrict__ out,
                           int N, int S) {
  __shared__ float dist[4096];
  __shared__ float rv[256];
  __shared__ int   ri[256];
  __shared__ int   far_s;
  int b = blockIdx.x;
  int tid = threadIdx.x;
  const float* P = xyz + (size_t)b * N * 3;
  for (int i = tid; i < N; i += 256) dist[i] = 1e10f;
  if (tid == 0) far_s = 0;
  __syncthreads();
  for (int s = 0; s < S; ++s) {
    int f = far_s;
    if (tid == 0) out[b * S + s] = f;
    float cx = P[f * 3 + 0], cy = P[f * 3 + 1], cz = P[f * 3 + 2];
    float best = -1.0f; int bi = N;
    for (int i = tid; i < N; i += 256) {
      float dx = P[i * 3 + 0] - cx, dy = P[i * 3 + 1] - cy, dz = P[i * 3 + 2] - cz;
      float d = dx * dx + dy * dy + dz * dz;
      float nd = fminf(dist[i], d);
      dist[i] = nd;
      if (nd > best) { best = nd; bi = i; }
    }
    rv[tid] = best; ri[tid] = bi;
    __syncthreads();
    for (int o = 128; o > 0; o >>= 1) {
      if (tid < o) {
        float ov = rv[tid + o]; int oi = ri[tid + o];
        if (ov > rv[tid] || (ov == rv[tid] && oi < ri[tid])) { rv[tid] = ov; ri[tid] = oi; }
      }
      __syncthreads();
    }
    if (tid == 0) far_s = ri[0];
    __syncthreads();
  }
}

// dst[b,s,c] = src[b, idx[b,s], c]
__global__ void gather_rows_kernel(const float* __restrict__ src, const int* __restrict__ idx,
                                   float* __restrict__ dst, int B, int Nsrc, int S, int C) {
  int t = blockIdx.x * blockDim.x + threadIdx.x;
  if (t >= B * S * C) return;
  int c = t % C, bs = t / C, b = bs / S;
  dst[t] = src[((size_t)b * Nsrc + idx[bs]) * C + c];
}

// ---------------------------------------------------------------------------
// Ball query: first NSAMPLE in-radius indices in ascending order, pad w/ first.
// ---------------------------------------------------------------------------
__global__ void ball_query_kernel(const float* __restrict__ xyz, const float* __restrict__ nxyz,
                                  int* __restrict__ idx, int B, int N, int S, float r2) {
  int t = blockIdx.x * blockDim.x + threadIdx.x;
  if (t >= B * S) return;
  int b = t / S;
  const float* P = xyz + (size_t)b * N * 3;
  const float* q = nxyz + (size_t)t * 3;
  int* o = idx + (size_t)t * NSAMPLE;
  int cnt = 0, first = 0;
  for (int j = 0; j < N; ++j) {
    float dx = P[j * 3 + 0] - q[0], dy = P[j * 3 + 1] - q[1], dz = P[j * 3 + 2] - q[2];
    float d = dx * dx + dy * dy + dz * dz;
    if (d <= r2) {
      if (cnt == 0) first = j;
      o[cnt++] = j;
      if (cnt == NSAMPLE) break;
    }
  }
  for (; cnt < NSAMPLE; ++cnt) o[cnt] = first;
}

// grouped row = [xyz[idx]-new_xyz | feats[idx] | zeros...] f16, stride Cpad
__global__ void group_kernel(const float* __restrict__ xyz, const float* __restrict__ feats,
                             const float* __restrict__ nxyz, const int* __restrict__ idx,
                             _Float16* __restrict__ out, int B, int N, int S, int Cf, int Cpad) {
  int t = blockIdx.x * blockDim.x + threadIdx.x;
  int total = B * S * NSAMPLE;
  if (t >= total) return;
  int bs = t / NSAMPLE;
  int b = bs / S;
  int j = idx[t];
  const float* P = xyz + ((size_t)b * N + j) * 3;
  const float* q = nxyz + (size_t)bs * 3;
  _Float16* o = out + (size_t)t * Cpad;
  o[0] = (_Float16)(P[0] - q[0]);
  o[1] = (_Float16)(P[1] - q[1]);
  o[2] = (_Float16)(P[2] - q[2]);
  const float* F = feats + ((size_t)b * N + j) * Cf;
  for (int c = 0; c < Cf; ++c) o[3 + c] = (_Float16)F[c];
  for (int c = 3 + Cf; c < Cpad; ++c) o[c] = (_Float16)0.0f;
}

// ---------------------------------------------------------------------------
// WMMA GEMM: Y(MxN,f16) = A(MxKpad,f16,row-stride lda) * W(KxN,f32) + bias.
// Block = 4 waves = 64x16 output tile at one n-tile. Weight panel (Kpad x 16)
// staged in LDS as f16 column-major; A loads are 2x b128 per k-step; D stores
// coalesced to b128 via per-wave LDS stage. Branch-free inner loop.
// Requires: M % 64 == 0, N % 16 == 0, lda % 32 == 0, Kpad % 32 == 0.
// ---------------------------------------------------------------------------
__global__ void wmma_gemm_bias(const _Float16* __restrict__ A, int lda,
                               const float* __restrict__ W,
                               const float* __restrict__ bias, _Float16* __restrict__ Y,
                               int K, int Kpad, int N) {
  extern __shared__ _Float16 lds[];
  _Float16* panel = lds;                        // [16][Kpad] column-major
  int tid = threadIdx.x;
  int n0 = blockIdx.y << 4;
  for (int i = tid; i < (Kpad << 4); i += blockDim.x) {
    int n = i & 15, k = i >> 4;
    panel[n * Kpad + k] = (k < K) ? (_Float16)W[(size_t)k * N + (n0 + n)] : (_Float16)0.0f;
  }
  __syncthreads();

  int wave = tid >> 5, lane = tid & 31;
  int l15 = lane & 15, hi = lane >> 4;
  int m0 = (blockIdx.x * 4 + wave) << 4;
  const _Float16* Arow = A + (size_t)(m0 + l15) * lda + hi * 8;
  const _Float16* Bcol = panel + l15 * Kpad + hi * 8;
  v8f acc = {};
  for (int k0 = 0; k0 < Kpad; k0 += 32) {
    v8h a0 = *(const v8h*)(Arow + k0);
    v8h a1 = *(const v8h*)(Arow + k0 + 16);
    v8h b0 = *(const v8h*)(Bcol + k0);
    v8h b1 = *(const v8h*)(Bcol + k0 + 16);
    v16h a = __builtin_shufflevector(a0, a1, 0,1,2,3,4,5,6,7,8,9,10,11,12,13,14,15);
    v16h b = __builtin_shufflevector(b0, b1, 0,1,2,3,4,5,6,7,8,9,10,11,12,13,14,15);
    acc = __builtin_amdgcn_wmma_f32_16x16x32_f16(false, a, false, b,
                                                 (short)0, acc, false, false);
  }

  // stage D (+bias, ->f16) in LDS, then coalesced b128 stores
  _Float16* stage = lds + (Kpad << 4) + (wave << 8);   // 256 halves per wave
  float bv = bias ? bias[n0 + l15] : 0.0f;
#pragma unroll
  for (int r = 0; r < 8; ++r)
    stage[((r + hi * 8) << 4) + l15] = (_Float16)(acc[r] + bv);
  __syncthreads();
  int srow = lane >> 1, scol = (lane & 1) << 3;
  v8h outv = *(const v8h*)(stage + (srow << 4) + scol);
  *(v8h*)(Y + (size_t)(m0 + srow) * N + n0 + scol) = outv;
}

__global__ void zero_f32_kernel(float* __restrict__ p, int n) {
  int t = blockIdx.x * blockDim.x + threadIdx.x;
  if (t < n) p[t] = 0.0f;
}

// per-channel sum & sumsq over M rows (block-reduce + atomicAdd)
__global__ void bn_stats_kernel(const _Float16* __restrict__ Y, float* __restrict__ stat,
                                int M, int N) {
  __shared__ float ss[256], sq[256];
  int c = blockIdx.x;
  int tid = threadIdx.x;
  float s = 0.f, q = 0.f;
  for (int m = blockIdx.y * 256 + tid; m < M; m += gridDim.y * 256) {
    float v = (float)Y[(size_t)m * N + c];
    s += v; q += v * v;
  }
  ss[tid] = s; sq[tid] = q;
  __syncthreads();
  for (int o = 128; o > 0; o >>= 1) {
    if (tid < o) { ss[tid] += ss[tid + o]; sq[tid] += sq[tid + o]; }
    __syncthreads();
  }
  if (tid == 0) { atomicAdd(&stat[c], ss[0]); atomicAdd(&stat[N + c], sq[0]); }
}

__global__ void bn_relu_kernel(_Float16* __restrict__ Y, const float* __restrict__ stat,
                               const float* __restrict__ g, const float* __restrict__ be,
                               int M, int N) {
  int t = blockIdx.x * blockDim.x + threadIdx.x;
  int total = M * N;
  if (t >= total) return;
  int c = t % N;
  float invM = 1.0f / (float)M;
  float mu = stat[c] * invM;
  float var = stat[N + c] * invM - mu * mu;
  float v = (float)Y[(size_t)t];
  v = (v - mu) * rsqrtf(var + BN_EPS) * g[c] + be[c];
  Y[(size_t)t] = (_Float16)(v > 0.f ? v : 0.f);
}

// max over nsample dim: (BS, K, C) f16 -> (BS, C) f32
__global__ void maxpool_kernel(const _Float16* __restrict__ Y, float* __restrict__ out,
                               int BS, int Kk, int C) {
  int t = blockIdx.x * blockDim.x + threadIdx.x;
  if (t >= BS * C) return;
  int c = t % C, bs = t / C;
  const _Float16* p = Y + (size_t)bs * Kk * C + c;
  float m = (float)p[0];
  for (int k = 1; k < Kk; ++k) m = fmaxf(m, (float)p[(size_t)k * C]);
  out[t] = m;
}

// 3-NN inverse-distance interp + concat -> f16 rows of stride Cpad
__global__ void fp_interp_kernel(const float* __restrict__ xyz1, const float* __restrict__ xyz2,
                                 const float* __restrict__ p1, int C1,
                                 const float* __restrict__ p2, int C2,
                                 _Float16* __restrict__ out, int B, int N1, int N2, int Cpad) {
  int t = blockIdx.x * blockDim.x + threadIdx.x;
  if (t >= B * N1) return;
  int b = t / N1;
  const float* q = xyz1 + (size_t)t * 3;
  const float* P2 = xyz2 + (size_t)b * N2 * 3;
  float d0 = 1e30f, d1 = 1e30f, d2 = 1e30f; int i0 = 0, i1 = 0, i2 = 0;
  for (int j = 0; j < N2; ++j) {
    float dx = P2[j * 3 + 0] - q[0], dy = P2[j * 3 + 1] - q[1], dz = P2[j * 3 + 2] - q[2];
    float d = dx * dx + dy * dy + dz * dz;
    if (d < d0)      { d2 = d1; i2 = i1; d1 = d0; i1 = i0; d0 = d; i0 = j; }
    else if (d < d1) { d2 = d1; i2 = i1; d1 = d;  i1 = j; }
    else if (d < d2) { d2 = d;  i2 = j; }
  }
  float w0 = 1.f / (d0 + 1e-8f), w1 = 1.f / (d1 + 1e-8f), w2 = 1.f / (d2 + 1e-8f);
  float ws = w0 + w1 + w2; w0 /= ws; w1 /= ws; w2 /= ws;
  _Float16* o = out + (size_t)t * Cpad;
  const float* f1p = p1 + (size_t)t * C1;
  for (int c = 0; c < C1; ++c) o[c] = (_Float16)f1p[c];
  const float* a0 = p2 + ((size_t)b * N2 + i0) * C2;
  const float* a1 = p2 + ((size_t)b * N2 + i1) * C2;
  const float* a2 = p2 + ((size_t)b * N2 + i2) * C2;
  for (int c = 0; c < C2; ++c)
    o[C1 + c] = (_Float16)(w0 * a0[c] + w1 * a1[c] + w2 * a2[c]);
  for (int c = C1 + C2; c < Cpad; ++c) o[c] = (_Float16)0.0f;
}

__global__ void h2f_kernel(const _Float16* __restrict__ s, float* __restrict__ d, int n) {
  int t = blockIdx.x * blockDim.x + threadIdx.x;
  if (t < n) d[t] = (float)s[t];
}

// conv2 (128 -> 13) + log_softmax, f32 output
__global__ void head_kernel(const _Float16* __restrict__ H, const float* __restrict__ W2,
                            const float* __restrict__ b2, float* __restrict__ out,
                            int M, int K, int C) {
  int t = blockIdx.x * blockDim.x + threadIdx.x;
  if (t >= M) return;
  const _Float16* h = H + (size_t)t * K;
  float logits[16];
  for (int j = 0; j < C; ++j) logits[j] = b2[j];
  for (int c = 0; c < K; ++c) {
    float hv = (float)h[c];
    const float* wr = W2 + (size_t)c * C;
    for (int j = 0; j < C; ++j) logits[j] += hv * wr[j];
  }
  float mx = logits[0];
  for (int j = 1; j < C; ++j) mx = fmaxf(mx, logits[j]);
  float se = 0.f;
  for (int j = 0; j < C; ++j) se += expf(logits[j] - mx);
  float ls = logf(se);
  float* o = out + (size_t)t * C;
  for (int j = 0; j < C; ++j) o[j] = logits[j] - mx - ls;
}

// ---------------------------------------------------------------------------
// Host-side orchestration
// ---------------------------------------------------------------------------
static void run_layer(const _Float16* X, int lda, _Float16* Y,
                      const float* W, const float* bias, const float* g, const float* be,
                      int M, int K, int N, float* stat, hipStream_t stream) {
  zero_f32_kernel<<<cdiv_i(2 * N, 256), 256, 0, stream>>>(stat, 2 * N);
  int Kpad = round32(K);
  dim3 grid((unsigned)(M / 64), (unsigned)(N / 16));
  size_t ldsBytes = (size_t)Kpad * 16 * 2 + 4 * 256 * 2;
  wmma_gemm_bias<<<grid, 128, ldsBytes, stream>>>(X, lda, W, bias, Y, K, Kpad, N);
  int gy = cdiv_i(M, 256); if (gy > 64) gy = 64;
  dim3 sg((unsigned)N, (unsigned)gy);
  bn_stats_kernel<<<sg, 256, 0, stream>>>(Y, stat, M, N);
  int total = M * N;
  bn_relu_kernel<<<cdiv_i(total, 256), 256, 0, stream>>>(Y, stat, g, be, M, N);
}

static const _Float16* run_mlp(void* const* d_in, int base, int nl, const int* dims,
                               int M, const _Float16* X, int lda0,
                               _Float16* bufA, _Float16* bufB,
                               float* stat, hipStream_t stream) {
  const _Float16* cur = X;
  int lda = lda0;
  for (int l = 0; l < nl; ++l) {
    _Float16* out = (l & 1) ? bufB : bufA;
    run_layer(cur, lda, out,
              (const float*)d_in[base + 4 * l + 0], (const float*)d_in[base + 4 * l + 1],
              (const float*)d_in[base + 4 * l + 2], (const float*)d_in[base + 4 * l + 3],
              M, dims[l], dims[l + 1], stat, stream);
    cur = out;
    lda = dims[l + 1];
  }
  return cur;
}

static void run_sa(void* const* d_in, int base, int nl, const int* dims,
                   const float* xyz_in, int Nin, const float* feats, int Cf,
                   int S, float radius, float* xyz_out, float* feat_out,
                   int* fpsidx, int* bqidx, _Float16* bufX, _Float16* bufA, _Float16* bufB,
                   float* stat, hipStream_t stream) {
  const int B = 8;
  fps_kernel<<<B, 256, 0, stream>>>(xyz_in, fpsidx, Nin, S);
  gather_rows_kernel<<<cdiv_i(B * S * 3, 256), 256, 0, stream>>>(xyz_in, fpsidx, xyz_out, B, Nin, S, 3);
  ball_query_kernel<<<cdiv_i(B * S, 128), 128, 0, stream>>>(xyz_in, xyz_out, bqidx, B, Nin, S, radius * radius);
  int M = B * S * NSAMPLE;
  int Cpad = round32(3 + Cf);
  group_kernel<<<cdiv_i(M, 256), 256, 0, stream>>>(xyz_in, feats, xyz_out, bqidx, bufX, B, Nin, S, Cf, Cpad);
  const _Float16* last = run_mlp(d_in, base, nl, dims, M, bufX, Cpad, bufA, bufB, stat, stream);
  int Cl = dims[nl];
  maxpool_kernel<<<cdiv_i(B * S * Cl, 256), 256, 0, stream>>>(last, feat_out, B * S, NSAMPLE, Cl);
}

static const _Float16* run_fp(void* const* d_in, int base, int nl, const int* dims,
                              const float* xyz1, int N1, const float* xyz2, int N2,
                              const float* p1, int C1, const float* p2, int C2,
                              _Float16* bufX, _Float16* bufA, _Float16* bufB,
                              float* stat, hipStream_t stream) {
  const int B = 8;
  int Cpad = round32(C1 + C2);
  fp_interp_kernel<<<cdiv_i(B * N1, 128), 128, 0, stream>>>(xyz1, xyz2, p1, C1, p2, C2,
                                                            bufX, B, N1, N2, Cpad);
  return run_mlp(d_in, base, nl, dims, B * N1, bufX, Cpad, bufA, bufB, stat, stream);
}

extern "C" void kernel_launch(void* const* d_in, const int* in_sizes, int n_in,
                              void* d_out, int out_size, void* d_ws, size_t ws_size,
                              hipStream_t stream) {
  (void)in_sizes; (void)n_in; (void)out_size; (void)ws_size;
  const int B = 8, N0 = 4096;

  // --- workspace carve ---
  char* wsb = (char*)d_ws;
  size_t off = 0;
  auto alloc = [&](size_t bytes) -> void* {
    void* p = wsb + off;
    off += (bytes + 255) & ~(size_t)255;
    return p;
  };
  float* xyz0 = (float*)alloc((size_t)B * N0 * 3 * 4);
  float* nor0 = (float*)alloc((size_t)B * N0 * 3 * 4);
  float* xyz1 = (float*)alloc((size_t)B * 1024 * 3 * 4);
  float* f1   = (float*)alloc((size_t)B * 1024 * 64 * 4);
  float* xyz2 = (float*)alloc((size_t)B * 256 * 3 * 4);
  float* f2   = (float*)alloc((size_t)B * 256 * 128 * 4);
  float* xyz3 = (float*)alloc((size_t)B * 64 * 3 * 4);
  float* f3   = (float*)alloc((size_t)B * 64 * 256 * 4);
  float* xyz4 = (float*)alloc((size_t)B * 16 * 3 * 4);
  float* f4   = (float*)alloc((size_t)B * 16 * 512 * 4);
  float* l3   = (float*)alloc((size_t)B * 64 * 256 * 4);
  float* l2   = (float*)alloc((size_t)B * 256 * 256 * 4);
  float* l1   = (float*)alloc((size_t)B * 1024 * 128 * 4);
  int*   fpsidx = (int*)alloc((size_t)B * 1024 * 4);
  int*   bqidx  = (int*)alloc((size_t)B * 1024 * NSAMPLE * 4);
  float* stat   = (float*)alloc(1024 * 4);
  _Float16* bufX = (_Float16*)alloc((size_t)13107200 * 2);  // max M*Cpad (sa1: 409600*32)
  _Float16* bufA = (_Float16*)alloc((size_t)26214400 * 2);  // max M*width (sa1: 409600*64)
  _Float16* bufB = (_Float16*)alloc((size_t)26214400 * 2);

  // --- param index map (setup_inputs insertion order, tuples flattened) ---
  const float* x = (const float*)d_in[0];
  const int sa1b = 1, sa2b = 13, sa3b = 25, sa4b = 37;
  const int fp1b = 49, fp2b = 57, fp3b = 65, fp4b = 73;
  const float* conv1_W = (const float*)d_in[85];
  const float* conv1_b = (const float*)d_in[86];
  const float* bn1_g   = (const float*)d_in[87];
  const float* bn1_b   = (const float*)d_in[88];
  const float* conv2_W = (const float*)d_in[89];
  const float* conv2_b = (const float*)d_in[90];

  const int sa1d[4] = {6, 32, 32, 64};
  const int sa2d[4] = {67, 64, 64, 128};
  const int sa3d[4] = {131, 128, 128, 256};
  const int sa4d[4] = {259, 256, 256, 512};
  const int fp1d[3] = {768, 256, 256};
  const int fp2d[3] = {384, 256, 256};
  const int fp3d[3] = {320, 256, 128};
  const int fp4d[4] = {131, 128, 128, 128};

  // --- split input ---
  split_kernel<<<cdiv_i(B * N0, 256), 256, 0, stream>>>(x, xyz0, nor0, B, N0);

  // --- set-abstraction stages ---
  run_sa(d_in, sa1b, 3, sa1d, xyz0, N0,   nor0, 3,   1024, 0.1f, xyz1, f1,
         fpsidx, bqidx, bufX, bufA, bufB, stat, stream);
  run_sa(d_in, sa2b, 3, sa2d, xyz1, 1024, f1,   64,  256,  0.2f, xyz2, f2,
         fpsidx, bqidx, bufX, bufA, bufB, stat, stream);
  run_sa(d_in, sa3b, 3, sa3d, xyz2, 256,  f2,   128, 64,   0.4f, xyz3, f3,
         fpsidx, bqidx, bufX, bufA, bufB, stat, stream);
  run_sa(d_in, sa4b, 3, sa4d, xyz3, 64,   f3,   256, 16,   0.8f, xyz4, f4,
         fpsidx, bqidx, bufX, bufA, bufB, stat, stream);

  // --- feature propagation stages ---
  const _Float16* r;
  r = run_fp(d_in, fp1b, 2, fp1d, xyz3, 64,   xyz4, 16,  f3,   256, f4, 512,
             bufX, bufA, bufB, stat, stream);
  h2f_kernel<<<cdiv_i(B * 64 * 256, 256), 256, 0, stream>>>(r, l3, B * 64 * 256);
  r = run_fp(d_in, fp2b, 2, fp2d, xyz2, 256,  xyz3, 64,  f2,   128, l3, 256,
             bufX, bufA, bufB, stat, stream);
  h2f_kernel<<<cdiv_i(B * 256 * 256, 256), 256, 0, stream>>>(r, l2, B * 256 * 256);
  r = run_fp(d_in, fp3b, 2, fp3d, xyz1, 1024, xyz2, 256, f1,   64,  l2, 256,
             bufX, bufA, bufB, stat, stream);
  h2f_kernel<<<cdiv_i(B * 1024 * 128, 256), 256, 0, stream>>>(r, l1, B * 1024 * 128);
  const _Float16* l0 = run_fp(d_in, fp4b, 3, fp4d, xyz0, N0, xyz1, 1024,
                              nor0, 3, l1, 128, bufX, bufA, bufB, stat, stream);
  // fp4 has 3 layers -> l0 == bufA (f16, B*4096 x 128)

  // --- conv1 + bn + relu (another WMMA layer), writes bufB ---
  run_layer(l0, 128, bufB, conv1_W, conv1_b, bn1_g, bn1_b, B * N0, 128, 128, stat, stream);

  // --- conv2 + log_softmax -> d_out ---
  head_kernel<<<cdiv_i(B * N0, 128), 128, 0, stream>>>(bufB, conv2_W, conv2_b,
                                                       (float*)d_out, B * N0, 128, 13);
}
